// CoverageAttention_66322884985219
// MI455X (gfx1250) — matrix operand order repaired and checked by reading, not compiled, not run on hardware
//
#include <hip/hip_runtime.h>
#include <hip/hip_bf16.h>
#include <math.h>

// ---------------------------------------------------------------------------
// CoverageAttention for MI455X (gfx1250): wave32 + v_wmma_f32_16x16x32_f16,
// async global->LDS staging (double-buffered), DPP16 row_xmask reductions.
// B=8, TQ=TK=1024, D_MODEL=1024, NHEAD=16, HEAD_DIM=64, SCALE=8
// Workspace use: ~90 MB.
// ---------------------------------------------------------------------------

typedef __attribute__((ext_vector_type(16))) _Float16 v16h;
typedef __attribute__((ext_vector_type(8)))  _Float16 v8h;
typedef __attribute__((ext_vector_type(4)))  _Float16 v4h;
typedef __attribute__((ext_vector_type(8)))  float    v8f;
typedef __attribute__((ext_vector_type(4)))  int      i32x4;

static constexpr int D_MODEL  = 1024;
static constexpr int NHEAD    = 16;
static constexpr int HEAD_DIM = 64;
static constexpr int BATCH    = 8;
static constexpr int TSEQ     = 1024;
static constexpr float INV_SCALE = 0.125f;       // 1/sqrt(64)
static constexpr float INV_H     = 1.0f / 16.0f;

// ---- async global->LDS staging (guarded; falls back to VGPR copy) ---------
#if defined(__has_builtin)
#if __has_builtin(__builtin_amdgcn_global_load_async_to_lds_b128)
#define ASYNC_STAGE 1
#endif
#endif
#ifndef ASYNC_STAGE
#define ASYNC_STAGE 0
#endif

__device__ __forceinline__ void stage_b128(const _Float16* g, _Float16* l) {
#if ASYNC_STAGE
  // (v4i addrspace(1)* src, v4i addrspace(3)* dst, imm offset, imm cpol)
  typedef __attribute__((address_space(1))) i32x4 g_v4i;
  typedef __attribute__((address_space(3))) i32x4 l_v4i;
  __builtin_amdgcn_global_load_async_to_lds_b128((g_v4i*)g, (l_v4i*)l, 0, 0);
#else
  *(v8h*)l = *(const v8h*)g;
#endif
}

__device__ __forceinline__ void wait_async_lds() {
#if ASYNC_STAGE
#if __has_builtin(__builtin_amdgcn_s_wait_asynccnt)
  __builtin_amdgcn_s_wait_asynccnt(0);
#else
  asm volatile("s_wait_asynccnt 0" ::: "memory");
#endif
#endif
}

// ---- WMMA wrapper (8-arg probe-confirmed signature) -----------------------
__device__ __forceinline__ v8f wmma_f16(v16h a, v16h b, v8f c) {
  return __builtin_amdgcn_wmma_f32_16x16x32_f16(
      /*neg_a=*/false, a, /*neg_b=*/false, b,
      /*c_mod=*/(short)0, c, /*reuse_a=*/false, /*reuse_b=*/false);
}

// A-fragment (16x32, MxK): lane holds row (lane&15), halves
// {kb..kb+7, kb+16..kb+23}, kb = (lane<16)?0:8.
__device__ __forceinline__ v16h load_a_frag(const _Float16* rowptr, int kb) {
  union { v16h v; v8h h[2]; } u;
  u.h[0] = *(const v8h*)(rowptr + kb);
  u.h[1] = *(const v8h*)(rowptr + kb + 16);
  return u.v;
}

// B-fragment (32x16, KxN): lane holds column (lane&15), halves
// {kb..kb+7, kb+16..kb+23}, kb = kb2/2 = (lane<16)?0:8.
__device__ __forceinline__ v16h load_b_frag(const _Float16* rowptr, int kb2) {
  union { v16h v; v8h h[2]; } u;
  u.h[0] = *(const v8h*)(rowptr + (kb2 >> 1));
  u.h[1] = *(const v8h*)(rowptr + (kb2 >> 1) + 16);
  return u.v;
}

// ---- 16-lane row reductions via DPP16 row_xmask (no LDS, no waits) --------
template <int CTRL>
__device__ __forceinline__ float dpp_xor_f32(float x) {
  return __int_as_float(__builtin_amdgcn_update_dpp(
      0, __float_as_int(x), CTRL, 0xf, 0xf, true));
}
__device__ __forceinline__ float rmax16(float v) {
  v = fmaxf(v, dpp_xor_f32<0x161>(v));
  v = fmaxf(v, dpp_xor_f32<0x162>(v));
  v = fmaxf(v, dpp_xor_f32<0x164>(v));
  v = fmaxf(v, dpp_xor_f32<0x168>(v));
  return v;
}
__device__ __forceinline__ float rsum16(float v) {
  v += dpp_xor_f32<0x161>(v);
  v += dpp_xor_f32<0x162>(v);
  v += dpp_xor_f32<0x164>(v);
  v += dpp_xor_f32<0x168>(v);
  return v;
}

// ---------------------------------------------------------------------------
// fp32 -> f16 bulk convert (one pass; removes all converts from GEMM loops)
// ---------------------------------------------------------------------------
__global__ __launch_bounds__(256) void cvt_f16_kernel(
    _Float16* __restrict__ dst, const float* __restrict__ src, int n4) {
  int i = blockIdx.x * 256 + threadIdx.x;
  if (i < n4) {
    float4 f = ((const float4*)src)[i];
    v4h h;
    h[0] = (_Float16)f.x; h[1] = (_Float16)f.y;
    h[2] = (_Float16)f.z; h[3] = (_Float16)f.w;
    ((v4h*)dst)[i] = h;
  }
}

// ---------------------------------------------------------------------------
// Kernel 1: f16 GEMM  Y[m,n] = sum_k A[m,k] * W[n,k] + bias[n]
// Double-buffered async LDS pipeline: per k-chunk, wait chunk i, barrier,
// issue async stage of chunk i+1, then WMMA on chunk i (copy under compute).
// Block: 128x128 tile, 256 thr = 8 waves, wave = 32x64 subtile.
// ---------------------------------------------------------------------------
template <bool OUT_F16>
__global__ __launch_bounds__(256) void wmma_gemm_f16_kernel(
    const _Float16* __restrict__ A, const _Float16* __restrict__ Wh,
    const float* __restrict__ bias, void* __restrict__ Out,
    int M, int N, int K) {
  constexpr int LDA = 40;  // halves per LDS row (bank-conflict padding)
  __shared__ __align__(16) _Float16 As[2][128 * LDA];
  __shared__ __align__(16) _Float16 Ws[2][128 * LDA];

  const int tid  = threadIdx.x;
  const int lane = tid & 31;
  const int wave = tid >> 5;
  const int wr   = wave & 3;   // 4 row groups of 32
  const int wc   = wave >> 2;  // 2 col groups of 64
  const int m0   = blockIdx.y * 128;
  const int n0   = blockIdx.x * 128;

  const int nl  = lane & 15;
  const int mb  = (lane < 16) ? 0 : 8;
  const int kbA = (lane < 16) ? 0 : 8;
  const int kbB = (lane < 16) ? 0 : 16;

  const _Float16* Ag = A  + (size_t)m0 * K;
  const _Float16* Wg = Wh + (size_t)n0 * K;

  auto stage = [&](int k0, int p) {
    // A and W tiles: 128 rows x 32 halves each = 512 b128 slots, 2/thread
    for (int s = tid; s < 512; s += 256) {
      const int r = s >> 2, c = (s & 3) << 3;
      stage_b128(Ag + (size_t)r * K + k0 + c, &As[p][r * LDA + c]);
      stage_b128(Wg + (size_t)r * K + k0 + c, &Ws[p][r * LDA + c]);
    }
    if (k0 + 32 < K) {  // hint one more chunk ahead into GL2
      const size_t po = (size_t)(tid >> 2) * K + k0 + 32 + ((tid & 3) << 3);
      __builtin_prefetch(Ag + po, 0, 1);
      __builtin_prefetch(Wg + po, 0, 1);
    }
  };

  v8f acc[2][4];
#pragma unroll
  for (int i = 0; i < 2; ++i)
#pragma unroll
    for (int j = 0; j < 4; ++j) acc[i][j] = {};

  stage(0, 0);
  const int nIter = K / 32;
  for (int it = 0; it < nIter; ++it) {
    const int p = it & 1;
    wait_async_lds();    // chunk `it` resident in LDS (my slots)
    __syncthreads();     // everyone's slots resident; buffer p^1 readers done
    if (it + 1 < nIter) stage((it + 1) * 32, p ^ 1);  // stream under compute

    v16h a[2];
#pragma unroll
    for (int mt = 0; mt < 2; ++mt)
      a[mt] = load_a_frag(&As[p][(wr * 32 + mt * 16 + nl) * LDA], kbA);
#pragma unroll
    for (int nt = 0; nt < 4; ++nt) {
      v16h b = load_b_frag(&Ws[p][(wc * 64 + nt * 16 + nl) * LDA], kbB);
      acc[0][nt] = wmma_f16(a[0], b, acc[0][nt]);
      acc[1][nt] = wmma_f16(a[1], b, acc[1][nt]);
    }
  }

  // ---- epilogue: C layout lane=(N=nl, M=mb+r) ----
#pragma unroll
  for (int mt = 0; mt < 2; ++mt) {
#pragma unroll
    for (int nt = 0; nt < 4; ++nt) {
      const int col = n0 + wc * 64 + nt * 16 + nl;
      const float bv = bias[col];
#pragma unroll
      for (int r = 0; r < 8; ++r) {
        const int row = m0 + wr * 32 + mt * 16 + mb + r;
        const float v = acc[mt][nt][r] + bv;
        if constexpr (OUT_F16)
          ((_Float16*)Out)[(size_t)row * N + col] = (_Float16)v;
        else
          ((float*)Out)[(size_t)row * N + col] = v;
      }
    }
  }
}

// ---------------------------------------------------------------------------
// Kernel 2: softmax stats.  s[q,k] = (Q.K)/8 + coverage[b,k]*Wcov[h]
// Double-buffered async K staging.  Grid: (TQ/64, B*H), 128 thr = 4 waves.
// ---------------------------------------------------------------------------
__global__ __launch_bounds__(128) void attn_ml_kernel(
    const _Float16* __restrict__ Qh, const _Float16* __restrict__ Kh,
    const float* __restrict__ coverage, const float* __restrict__ Wcov,
    float* __restrict__ ml) {
  constexpr int LDK = 72;
  __shared__ float covb[TSEQ];
  __shared__ __align__(16) _Float16 Ks[2][32 * LDK];

  const int tid  = threadIdx.x;
  const int lane = tid & 31;
  const int wave = tid >> 5;
  const int b = blockIdx.y >> 4, h = blockIdx.y & 15;
  const int qbase = blockIdx.x * 64 + wave * 16;

  const float wch = Wcov[h];
  for (int i = tid; i < TSEQ; i += 128) covb[i] = coverage[b * TSEQ + i] * wch;

  const int nl  = lane & 15;
  const int mb  = (lane < 16) ? 0 : 8;
  const int kbA = (lane < 16) ? 0 : 8;
  const int kbB = (lane < 16) ? 0 : 16;

  const _Float16* qptr =
      Qh + ((size_t)(b * TSEQ + qbase + nl)) * D_MODEL + h * HEAD_DIM;
  const v16h qa0 = load_a_frag(qptr, kbA);
  const v16h qa1 = load_a_frag(qptr + 32, kbA);

  auto stageK = [&](int k0, int p) {
    const _Float16* Kg =
        Kh + ((size_t)(b * TSEQ + k0)) * D_MODEL + h * HEAD_DIM;
    for (int s = tid; s < 256; s += 128) {
      const int r = s >> 3, c = (s & 7) << 3;
      stage_b128(Kg + (size_t)r * D_MODEL + c, &Ks[p][r * LDK + c]);
    }
  };

  float m_run[8], l_run[8];
#pragma unroll
  for (int r = 0; r < 8; ++r) { m_run[r] = -__builtin_inff(); l_run[r] = 0.f; }

  stageK(0, 0);
  const int nIter = TSEQ / 32;
  for (int it = 0; it < nIter; ++it) {
    const int p = it & 1;
    const int k0 = it * 32;
    wait_async_lds();
    __syncthreads();
    if (it + 1 < nIter) stageK(k0 + 32, p ^ 1);

#pragma unroll
    for (int kt = 0; kt < 2; ++kt) {
      const _Float16* kr = &Ks[p][(kt * 16 + nl) * LDK];
      v16h b0 = load_b_frag(kr, kbB);       // d 0..31
      v16h b1 = load_b_frag(kr + 32, kbB);  // d 32..63
      v8f ct = {};
      ct = wmma_f16(qa0, b0, ct);
      ct = wmma_f16(qa1, b1, ct);
#pragma unroll
      for (int r = 0; r < 8; ++r) {
        const float s  = ct[r] * INV_SCALE + covb[k0 + kt * 16 + nl];
        const float mx = rmax16(s);
        const float mn = fmaxf(m_run[r], mx);
        const float es = rsum16(__expf(s - mn));
        l_run[r] = l_run[r] * __expf(m_run[r] - mn) + es;
        m_run[r] = mn;
      }
    }
  }

  if (nl == 0) {
#pragma unroll
    for (int r = 0; r < 8; ++r) {
      const size_t idx = ((size_t)(b * NHEAD + h)) * TSEQ + qbase + mb + r;
      ml[idx * 2 + 0] = m_run[r];
      ml[idx * 2 + 1] = l_run[r];
    }
  }
}

// ---------------------------------------------------------------------------
// Kernel 3: attention pass.  attn = exp(s-m)/l ; O = attn @ V ;
// covacc[b,k] += (1/H) * sum_q attn.  Double-buffered K/V staging.
// ---------------------------------------------------------------------------
__global__ __launch_bounds__(128) void attn_av_kernel(
    const _Float16* __restrict__ Qh, const _Float16* __restrict__ Kh,
    const _Float16* __restrict__ Vh, const float* __restrict__ coverage,
    const float* __restrict__ Wcov, const float* __restrict__ ml,
    _Float16* __restrict__ Ctx, float* __restrict__ covacc) {
  constexpr int LDK = 72, LDV = 40, LDP = 40;
  __shared__ float covb[TSEQ];
  __shared__ __align__(16) _Float16 Ks[2][32 * LDK];
  __shared__ __align__(16) _Float16 Vst[2][64 * LDV];  // transposed: [d][k]
  __shared__ __align__(16) _Float16 Ps[4][16 * LDP];   // per-wave P tile

  const int tid  = threadIdx.x;
  const int lane = tid & 31;
  const int wave = tid >> 5;
  const int b = blockIdx.y >> 4, h = blockIdx.y & 15;
  const int qbase = blockIdx.x * 64 + wave * 16;

  const float wch = Wcov[h];
  for (int i = tid; i < TSEQ; i += 128) covb[i] = coverage[b * TSEQ + i] * wch;

  const int nl  = lane & 15;
  const int mb  = (lane < 16) ? 0 : 8;
  const int kbA = (lane < 16) ? 0 : 8;
  const int kbB = (lane < 16) ? 0 : 16;

  const _Float16* qptr =
      Qh + ((size_t)(b * TSEQ + qbase + nl)) * D_MODEL + h * HEAD_DIM;
  const v16h qa0 = load_a_frag(qptr, kbA);
  const v16h qa1 = load_a_frag(qptr + 32, kbA);

  float mrow[8], linv[8];
#pragma unroll
  for (int r = 0; r < 8; ++r) {
    const size_t idx = ((size_t)(b * NHEAD + h)) * TSEQ + qbase + mb + r;
    mrow[r] = ml[idx * 2 + 0];
    linv[r] = 1.0f / ml[idx * 2 + 1];
  }

  auto stageKV = [&](int k0, int p) {
    const _Float16* Kg =
        Kh + ((size_t)(b * TSEQ + k0)) * D_MODEL + h * HEAD_DIM;
    const _Float16* Vg =
        Vh + ((size_t)(b * TSEQ + k0)) * D_MODEL + h * HEAD_DIM;
    for (int s = tid; s < 256; s += 128) {
      const int r = s >> 3, c = (s & 7) << 3;
      stage_b128(Kg + (size_t)r * D_MODEL + c, &Ks[p][r * LDK + c]);
      v8h hv = *(const v8h*)(Vg + (size_t)r * D_MODEL + c);
#pragma unroll
      for (int j = 0; j < 8; ++j) Vst[p][(c + j) * LDV + r] = hv[j];
    }
    if (k0 + 32 < TSEQ)
      __builtin_prefetch(
          Vg + 32 * D_MODEL + (size_t)(tid >> 3) * D_MODEL + ((tid & 7) << 3),
          0, 1);
  };

  v8f o[4];
#pragma unroll
  for (int dt = 0; dt < 4; ++dt) o[dt] = {};

  stageKV(0, 0);
  const int nIter = TSEQ / 32;
  for (int it = 0; it < nIter; ++it) {
    const int p = it & 1;
    const int k0 = it * 32;
    wait_async_lds();
    __syncthreads();
    if (it + 1 < nIter) stageKV(k0 + 32, p ^ 1);

#pragma unroll
    for (int kt = 0; kt < 2; ++kt) {
      const _Float16* kr = &Ks[p][(kt * 16 + nl) * LDK];
      v16h b0 = load_b_frag(kr, kbB);
      v16h b1 = load_b_frag(kr + 32, kbB);
      v8f ct = {};
      ct = wmma_f16(qa0, b0, ct);
      ct = wmma_f16(qa1, b1, ct);

      float cs = 0.f;
#pragma unroll
      for (int r = 0; r < 8; ++r) {
        const float s = ct[r] * INV_SCALE + covb[k0 + kt * 16 + nl];
        const float pr = __expf(s - mrow[r]) * linv[r];  // final attn prob
        cs += pr;
        Ps[wave][(mb + r) * LDP + kt * 16 + nl] = (_Float16)pr;
      }
      // column sum across both row groups (lane ^ 16 holds same column)
      cs += __shfl_xor(cs, 16, 32);
      if (lane < 16)
        atomicAdd(covacc + b * TSEQ + k0 + kt * 16 + nl, cs * INV_H);
    }

    // LDS is in-order per wave; make the P round-trip explicit anyway.
    asm volatile("s_wait_dscnt 0" ::: "memory");

    const v16h pa = load_a_frag(&Ps[wave][nl * LDP], kbA);  // P as A-operand
#pragma unroll
    for (int dt = 0; dt < 4; ++dt) {
      v16h bv = load_b_frag(&Vst[p][(dt * 16 + nl) * LDV], kbB);
      o[dt] = wmma_f16(pa, bv, o[dt]);
    }
  }

  // store context tile (already normalized): Ctx[b, q, h*64 + d] as f16
#pragma unroll
  for (int dt = 0; dt < 4; ++dt) {
#pragma unroll
    for (int r = 0; r < 8; ++r) {
      const int q = qbase + mb + r;
      Ctx[((size_t)(b * TSEQ + q)) * D_MODEL + h * HEAD_DIM + dt * 16 + nl] =
          (_Float16)o[dt][r];
    }
  }
}

// ---------------------------------------------------------------------------
// Tiny helpers: coverage init / final copy
// ---------------------------------------------------------------------------
__global__ void cov_init_kernel(float* __restrict__ covacc,
                                const float* __restrict__ coverage, int n) {
  int i = blockIdx.x * blockDim.x + threadIdx.x;
  if (i < n) covacc[i] = coverage[i];
}
__global__ void cov_copy_kernel(float* __restrict__ dst,
                                const float* __restrict__ src, int n) {
  int i = blockIdx.x * blockDim.x + threadIdx.x;
  if (i < n) dst[i] = src[i];
}

// ---------------------------------------------------------------------------
// Launch
// ---------------------------------------------------------------------------
extern "C" void kernel_launch(void* const* d_in, const int* in_sizes, int n_in,
                              void* d_out, int out_size, void* d_ws,
                              size_t ws_size, hipStream_t stream) {
  (void)in_sizes; (void)n_in; (void)out_size; (void)ws_size;

  const float* query    = (const float*)d_in[0];
  const float* memory   = (const float*)d_in[1];
  const float* coverage = (const float*)d_in[2];
  const float* Wq = (const float*)d_in[3];
  const float* bq = (const float*)d_in[4];
  const float* Wk = (const float*)d_in[5];
  const float* bk = (const float*)d_in[6];
  const float* Wv = (const float*)d_in[7];
  const float* bv = (const float*)d_in[8];
  const float* Wo = (const float*)d_in[9];
  const float* bo = (const float*)d_in[10];
  const float* Wcov = (const float*)d_in[11];

  const int M = BATCH * TSEQ;  // 8192
  const int N = D_MODEL, K = D_MODEL;

  // workspace layout (~90 MB)
  const size_t QKV = (size_t)M * D_MODEL * sizeof(_Float16);  // 16 MB
  const size_t MLB = (size_t)BATCH * NHEAD * TSEQ * 2 * sizeof(float);  // 1 MB
  const size_t CVB = (size_t)BATCH * TSEQ * sizeof(float);    // 32 KB
  const size_t WB  = (size_t)D_MODEL * D_MODEL * sizeof(_Float16);  // 2 MB
  char* ws = (char*)d_ws;
  _Float16* Qh   = (_Float16*)(ws + 0 * QKV);
  _Float16* Kh   = (_Float16*)(ws + 1 * QKV);
  _Float16* Vh   = (_Float16*)(ws + 2 * QKV);
  _Float16* Ctx  = (_Float16*)(ws + 3 * QKV);  // also Xq (dead before Ctx use)
  _Float16* Xq   = Ctx;                        // query f16 (projections only)
  _Float16* Xm   = (_Float16*)(ws + 4 * QKV);  // memory f16
  float*    ml   = (float*)(ws + 5 * QKV);
  float* covacc  = (float*)(ws + 5 * QKV + MLB);
  _Float16* W16  = (_Float16*)(ws + 5 * QKV + MLB + CVB);
  _Float16* Wq16 = W16 + 0 * (WB / sizeof(_Float16));
  _Float16* Wk16 = W16 + 1 * (WB / sizeof(_Float16));
  _Float16* Wv16 = W16 + 2 * (WB / sizeof(_Float16));
  _Float16* Wo16 = W16 + 3 * (WB / sizeof(_Float16));

  const int n4x  = M * D_MODEL / 4;        // activations, in float4s
  const int n4w  = D_MODEL * D_MODEL / 4;  // weights, in float4s
  const dim3 cG((n4x + 255) / 256), cW((n4w + 255) / 256);
  const dim3 gGrid(N / 128, M / 128);          // (8, 64)
  const dim3 aGrid(TSEQ / 64, BATCH * NHEAD);  // (16, 128)

  // 0: one-shot fp32 -> f16 conversion of all GEMM operands
  cvt_f16_kernel<<<cG, 256, 0, stream>>>(Xq, query, n4x);
  cvt_f16_kernel<<<cG, 256, 0, stream>>>(Xm, memory, n4x);
  cvt_f16_kernel<<<cW, 256, 0, stream>>>(Wq16, Wq, n4w);
  cvt_f16_kernel<<<cW, 256, 0, stream>>>(Wk16, Wk, n4w);
  cvt_f16_kernel<<<cW, 256, 0, stream>>>(Wv16, Wv, n4w);
  cvt_f16_kernel<<<cW, 256, 0, stream>>>(Wo16, Wo, n4w);

  // 1-3: Q/K/V projections (all-f16 async-pipelined GEMM)
  wmma_gemm_f16_kernel<true><<<gGrid, 256, 0, stream>>>(Xq, Wq16, bq, Qh, M, N, K);
  wmma_gemm_f16_kernel<true><<<gGrid, 256, 0, stream>>>(Xm, Wk16, bk, Kh, M, N, K);
  wmma_gemm_f16_kernel<true><<<gGrid, 256, 0, stream>>>(Xm, Wv16, bv, Vh, M, N, K);

  // 4: coverage accumulator = coverage
  cov_init_kernel<<<(BATCH * TSEQ + 255) / 256, 256, 0, stream>>>(
      covacc, coverage, BATCH * TSEQ);

  // 5: softmax stats
  attn_ml_kernel<<<aGrid, 128, 0, stream>>>(Qh, Kh, coverage, Wcov, ml);

  // 6: attention + coverage accumulation + context (overwrites Xq region)
  attn_av_kernel<<<aGrid, 128, 0, stream>>>(Qh, Kh, Vh, coverage, Wcov, ml,
                                            Ctx, covacc);

  // 7: output projection (f16 -> fp32 + bias) straight into d_out
  wmma_gemm_f16_kernel<false><<<gGrid, 256, 0, stream>>>(Ctx, Wo16, bo,
                                                         (float*)d_out, M, N, K);

  // 8: new_coverage appended after the [B,TQ,D] output
  float* cov_out = (float*)d_out + (size_t)M * D_MODEL;
  cov_copy_kernel<<<(BATCH * TSEQ + 255) / 256, 256, 0, stream>>>(
      cov_out, covacc, BATCH * TSEQ);
}
